// MYNET_85882166051372
// MI455X (gfx1250) — matrix operand order repaired and checked by reading, not compiled, not run on hardware
//
#include <hip/hip_runtime.h>
#include <math.h>

// ---------------------------------------------------------------------------
// Sizes (compile-time constants from the reference)
// ---------------------------------------------------------------------------
#define D      1024
#define H      512
#define SEQ    2048
#define BATCH  128
#define ENC    256
#define LSPAN  32          // min(MIN_SPAN*4, SEQ)
#define GSTEP  64          // SEQ/32
#define EPS    1e-8f

typedef __attribute__((ext_vector_type(16))) __bf16        v16bf;
typedef __attribute__((ext_vector_type(8)))  float         v8f;
typedef __attribute__((ext_vector_type(8)))  unsigned int  v8u;

__device__ __forceinline__ unsigned int pack_bf16(float lo, float hi)
{
    unsigned short a = __builtin_bit_cast(unsigned short, (__bf16)lo);
    unsigned short b = __builtin_bit_cast(unsigned short, (__bf16)hi);
    return (unsigned int)a | ((unsigned int)b << 16);
}

// ---------------------------------------------------------------------------
// Generic WMMA bf16 GEMM:  out[M,N] = epilogue( A[M,K] @ B[K,N] )
//   A rows addressed as  Abase + (r>>7)*lstride + (r&127)*rowstride
//   (lets the same kernel walk strided slices of long_x for the pools).
//   Block = 256 threads = 8 waves; block tile = 128 rows x 32 cols:
//   each wave owns 16 rows and multiplies ONE A fragment against TWO B
//   sub-tiles (A reuse), K staged 64-wide through LDS as packed-bf16 dword
//   pairs in exact WMMA VGPR-pair order (fragments = ds_load_b128 only).
//   4 independent accumulators -> 4 back-to-back v_wmma, no RAW hazards.
//   epilogue: v = acc (+ bias[n]) (+ addmat[(r&127)*N + n]); act(v);
//             (* mult[r*N+n]);  act: 0=none 1=gelu(exact) 2=sigmoid
// ---------------------------------------------------------------------------
__global__ __launch_bounds__(256)
void wmma_gemm_bf16(const float* __restrict__ Abase, long long lstride, int rowstride,
                    const float* __restrict__ B, int K, int N,
                    const float* __restrict__ bias,
                    const float* __restrict__ addmat,
                    const float* __restrict__ mult,
                    float* __restrict__ out, int act)
{
    // packed bf16 pairs: Asp[r][kp] = {A[r][2kp], A[r][2kp+1]}; 32 pairs + pad
    __shared__ __align__(16) unsigned int Asp[128][36];   // 144 B row stride
    __shared__ __align__(16) unsigned int Bsp[32][36];    // Bsp[n][kp] packed

    const int tid   = threadIdx.x;
    const int wave  = tid >> 5;          // 0..7  -> which 16-row slice
    const int lane  = tid & 31;
    const int m     = lane & 15;         // A row within tile / B column
    const int hh    = lane >> 4;         // lane half
    const int ncol0 = blockIdx.x * 32;
    const int mrow0 = blockIdx.y * 128;

    v8f acc00 = {}, acc01 = {}, acc10 = {}, acc11 = {};

    for (int k0 = 0; k0 < K; k0 += 64) {
        // --- A tile: 128 rows x 32 pairs; float2 global loads, pack to bf16 ---
        #pragma unroll
        for (int i = 0; i < 16; ++i) {
            int idx = tid + i * 256;                 // 0..4095
            int r   = idx >> 5;
            int kp  = idx & 31;
            int gr  = mrow0 + r;
            const float* ap = Abase + (long long)(gr >> 7) * lstride
                                    + (long long)(gr & 127) * rowstride
                                    + k0 + 2 * kp;
            float2 f = *(const float2*)ap;
            Asp[r][kp] = pack_bf16(f.x, f.y);
        }
        // --- B tile: 64 k x 32 n -> 1024 pairs, transposed-packed per column ---
        #pragma unroll
        for (int i = 0; i < 4; ++i) {
            int idx = tid + i * 256;                 // 0..1023
            int kp  = idx >> 5;                      // 0..31
            int n   = idx & 31;
            float b0e = B[(long long)(k0 + 2 * kp)     * N + ncol0 + n];
            float b1e = B[(long long)(k0 + 2 * kp + 1) * N + ncol0 + n];
            Bsp[n][kp] = pack_bf16(b0e, b1e);
        }
        __syncthreads();

        // --- two K=32 chunks x two N sub-tiles, 4 independent accumulators ---
        const int arow = wave * 16 + m;
        #pragma unroll
        for (int c = 0; c < 2; ++c) {
            const int off = c * 16;
            // A frag (ISA 16-bit A 16x32): VGPR j<4 -> pair 4h+j ; j>=4 -> 8+4h+(j-4)
            uint4 a0 = *(const uint4*)&Asp[arow][off + 4 * hh];
            uint4 a1 = *(const uint4*)&Asp[arow][off + 8 + 4 * hh];
            v8u au = { a0.x, a0.y, a0.z, a0.w, a1.x, a1.y, a1.z, a1.w };
            v16bf a = __builtin_bit_cast(v16bf, au);
            #pragma unroll
            for (int t = 0; t < 2; ++t) {
                // B frag (16-bit B 32x16): VGPR j<4 -> pair 4h+j ; j>=4 -> 8+4h+(j-4)
                uint4 b0 = *(const uint4*)&Bsp[t * 16 + m][off + 4 * hh];
                uint4 b1 = *(const uint4*)&Bsp[t * 16 + m][off + 8 + 4 * hh];
                v8u bu = { b0.x, b0.y, b0.z, b0.w, b1.x, b1.y, b1.z, b1.w };
                v16bf b = __builtin_bit_cast(v16bf, bu);
                if (c == 0 && t == 0)
                    acc00 = __builtin_amdgcn_wmma_f32_16x16x32_bf16(
                                false, a, false, b, (short)0, acc00, false, false);
                else if (c == 0 && t == 1)
                    acc01 = __builtin_amdgcn_wmma_f32_16x16x32_bf16(
                                false, a, false, b, (short)0, acc01, false, false);
                else if (c == 1 && t == 0)
                    acc10 = __builtin_amdgcn_wmma_f32_16x16x32_bf16(
                                false, a, false, b, (short)0, acc10, false, false);
                else
                    acc11 = __builtin_amdgcn_wmma_f32_16x16x32_bf16(
                                false, a, false, b, (short)0, acc11, false, false);
            }
        }
        __syncthreads();
    }

    // --- epilogue: C/D layout VGPR v -> row (v + 8*hh), col = lane&15 ---
    #pragma unroll
    for (int t = 0; t < 2; ++t) {
        v8f acc = (t == 0) ? (acc00 + acc10) : (acc01 + acc11);
        const int n = ncol0 + t * 16 + m;
        const float bv = bias ? bias[n] : 0.0f;
        #pragma unroll
        for (int v = 0; v < 8; ++v) {
            int gr = mrow0 + wave * 16 + v + 8 * hh;
            float val = acc[v] + bv;
            if (addmat) val += addmat[(gr & 127) * N + n];
            if (act == 1)      val = 0.5f * val * (1.0f + erff(val * 0.70710678118f));
            else if (act == 2) val = 1.0f / (1.0f + expf(-val));
            if (mult)   val *= mult[(long long)gr * N + n];
            out[(long long)gr * N + n] = val;
        }
    }
}

// ---------------------------------------------------------------------------
// ctx[b*D+d..d+3] = mean over ENC of encoded_x[e,b,d..d+3]   (float4)
// ---------------------------------------------------------------------------
__global__ void ctx_mean_kernel(const float* __restrict__ enc, float* __restrict__ ctx)
{
    int i = blockIdx.x * blockDim.x + threadIdx.x;     // 0..BATCH*D/4-1
    const float4* ep = (const float4*)enc + i;
    float4 s = {0.f, 0.f, 0.f, 0.f};
    for (int e = 0; e < ENC; ++e) {
        float4 v = ep[(long long)e * (BATCH * D / 4)];
        s.x += v.x; s.y += v.y; s.z += v.z; s.w += v.w;
    }
    const float inv = 1.0f / (float)ENC;
    float4 r = { s.x * inv, s.y * inv, s.z * inv, s.w * inv };
    ((float4*)ctx)[i] = r;
}

// span weights: sw[b] = softmax(hsp[b,:] @ sp_w2 + sp_b2)  (2 logits)
__global__ void span_kernel(const float* __restrict__ hsp,
                            const float* __restrict__ sp_w2,
                            const float* __restrict__ sp_b2,
                            float* __restrict__ sw)
{
    int b = blockIdx.x * blockDim.x + threadIdx.x;
    if (b >= BATCH) return;
    float s0 = sp_b2[0], s1 = sp_b2[1];
    const float* hp = hsp + (long long)b * H;
    for (int h = 0; h < H; ++h) {
        float v = hp[h];
        s0 += v * sp_w2[h * 2 + 0];
        s1 += v * sp_w2[h * 2 + 1];
    }
    float mx = fmaxf(s0, s1);
    float e0 = expf(s0 - mx), e1 = expf(s1 - mx);
    float inv = 1.0f / (e0 + e1);
    sw[b * 2 + 0] = e0 * inv;
    sw[b * 2 + 1] = e1 * inv;
}

// per-pool: score_l = sigmoid(h[l,b,:] . rs_w2 + rs_b2);
// w[l,b] = softmax_l(score + log(decay^(L-1-l)+eps))   (wave32 shuffles, L==32)
__global__ void weights_kernel(const float* __restrict__ h,
                               const float* __restrict__ rs_w2,
                               const float* __restrict__ rs_b2,
                               const float* __restrict__ decay_p,
                               float* __restrict__ wout)
{
    int b = blockIdx.x;          // 0..127
    int l = threadIdx.x;         // 0..31  (one wave32)
    const float4* hp = (const float4*)(h + ((long long)l * BATCH + b) * H);
    const float4* wp = (const float4*)rs_w2;
    float dot = 0.0f;
    for (int i = 0; i < H / 4; ++i) {
        float4 hv = hp[i], wv = wp[i];
        dot += hv.x * wv.x + hv.y * wv.y + hv.z * wv.z + hv.w * wv.w;
    }
    float score = 1.0f / (1.0f + expf(-(dot + rs_b2[0])));
    float decay = *decay_p;
    float logit = score + logf(powf(decay, (float)(LSPAN - 1 - l)) + EPS);

    float mx = logit;
    for (int off = 16; off > 0; off >>= 1)
        mx = fmaxf(mx, __shfl_xor(mx, off, 32));
    float e = expf(logit - mx);
    float s = e;
    for (int off = 16; off > 0; off >>= 1)
        s += __shfl_xor(s, off, 32);
    wout[l * BATCH + b] = e / s;
}

// pooled features + span scaling -> combined[b, 0:D]=local, [D:2D]=global (float4)
__global__ void pool_combine_kernel(const float* __restrict__ long_x,
                                    const float* __restrict__ wl,
                                    const float* __restrict__ wg,
                                    const float* __restrict__ sw,
                                    float* __restrict__ combined)
{
    int i = blockIdx.x * blockDim.x + threadIdx.x;     // 0..BATCH*D/4-1
    int b  = i >> 8;                                   // D/4 = 256 vecs per b
    int dv = i & 255;
    const float4* lx = (const float4*)long_x;
    float4 sl = {0.f,0.f,0.f,0.f}, sg = {0.f,0.f,0.f,0.f};
    #pragma unroll 4
    for (int l = 0; l < LSPAN; ++l) {
        float wlv = wl[l * BATCH + b];
        float wgv = wg[l * BATCH + b];
        float4 vl = lx[((long long)(SEQ - LSPAN + l) * BATCH + b) * (D / 4) + dv];
        float4 vg = lx[((long long)(l * GSTEP)       * BATCH + b) * (D / 4) + dv];
        sl.x += vl.x * wlv; sl.y += vl.y * wlv; sl.z += vl.z * wlv; sl.w += vl.w * wlv;
        sg.x += vg.x * wgv; sg.y += vg.y * wgv; sg.z += vg.z * wgv; sg.w += vg.w * wgv;
    }
    float s0 = sw[b * 2 + 0], s1 = sw[b * 2 + 1];
    float4 cl = { sl.x * s0, sl.y * s0, sl.z * s0, sl.w * s0 };
    float4 cg = { sg.x * s1, sg.y * s1, sg.z * s1, sg.w * s1 };
    ((float4*)combined)[(long long)b * (2 * D / 4) + dv]           = cl;
    ((float4*)combined)[(long long)b * (2 * D / 4) + (D / 4) + dv] = cg;
}

__global__ void fill_kernel(float4* __restrict__ out, int n4, float v)
{
    int i = blockIdx.x * blockDim.x + threadIdx.x;
    if (i < n4) { float4 f = {v, v, v, v}; out[i] = f; }
}

// ---------------------------------------------------------------------------
extern "C" void kernel_launch(void* const* d_in, const int* in_sizes, int n_in,
                              void* d_out, int out_size, void* d_ws, size_t ws_size,
                              hipStream_t stream)
{
    const float* long_x  = (const float*)d_in[0];
    const float* enc_x   = (const float*)d_in[1];
    const float* sp_w1   = (const float*)d_in[2];
    const float* sp_b1   = (const float*)d_in[3];
    const float* sp_w2   = (const float*)d_in[4];
    const float* sp_b2   = (const float*)d_in[5];
    const float* rs_w1   = (const float*)d_in[6];
    const float* rs_b1   = (const float*)d_in[7];
    const float* rs_w2   = (const float*)d_in[8];
    const float* rs_b2   = (const float*)d_in[9];
    const float* dec_l   = (const float*)d_in[10];
    const float* dec_g   = (const float*)d_in[11];
    const float* fu_w    = (const float*)d_in[12];
    const float* fu_b    = (const float*)d_in[13];
    const float* gate_w  = (const float*)d_in[14];
    const float* gate_b  = (const float*)d_in[15];

    float* out = (float*)d_out;

    // workspace layout (floats)
    float* ws       = (float*)d_ws;
    float* ctx      = ws;                      // 128*1024
    float* hsp      = ctx      + BATCH * D;    // 128*512
    float* hctx     = hsp      + BATCH * H;    // 128*512
    float* sw       = hctx     + BATCH * H;    // 128*2
    float* wl       = sw       + BATCH * 2;    // 32*128
    float* wg       = wl       + LSPAN * BATCH;// 32*128
    float* combined = wg       + LSPAN * BATCH;// 128*2048
    float* fusedbuf = combined + BATCH * 2 * D;// 128*1024
    float* hbuf     = fusedbuf + BATCH * D;    // 32*128*512 (reused by both pools)

    // 1) attention_weights = 1/SEQ everywhere (first SEQ*BATCH outputs)
    fill_kernel<<<(SEQ * BATCH / 4 + 255) / 256, 256, 0, stream>>>(
        (float4*)out, SEQ * BATCH / 4, 1.0f / (float)SEQ);

    // 2) ctx = mean(encoded_x, axis=0)
    ctx_mean_kernel<<<(BATCH * D / 4) / 256, 256, 0, stream>>>(enc_x, ctx);

    // 3) hsp = gelu(ctx @ sp_w1 + sp_b1)            M=128 K=1024 N=512
    wmma_gemm_bf16<<<dim3(H / 32, 1), 256, 0, stream>>>(
        ctx, 0LL, D, sp_w1, D, H, sp_b1, nullptr, nullptr, hsp, 1);

    // 4) span weights = softmax(hsp @ sp_w2 + sp_b2)
    span_kernel<<<1, 128, 0, stream>>>(hsp, sp_w2, sp_b2, sw);

    // 5) hctx = ctx @ rs_w1[D:2D,:] + rs_b1         (shared half of pool MLP)
    wmma_gemm_bf16<<<dim3(H / 32, 1), 256, 0, stream>>>(
        ctx, 0LL, D, rs_w1 + (long long)D * H, D, H, rs_b1, nullptr, nullptr, hctx, 0);

    // 6) local pool: h = gelu(hist @ rs_w1[:D,:] + hctx)   M=32*128 K=1024 N=512
    wmma_gemm_bf16<<<dim3(H / 32, (LSPAN * BATCH) / 128), 256, 0, stream>>>(
        long_x + (long long)(SEQ - LSPAN) * BATCH * D, (long long)BATCH * D, D,
        rs_w1, D, H, nullptr, hctx, nullptr, hbuf, 1);
    weights_kernel<<<BATCH, 32, 0, stream>>>(hbuf, rs_w2, rs_b2, dec_l, wl);

    // 7) global pool (stride-64 rows of long_x), reuse hbuf
    wmma_gemm_bf16<<<dim3(H / 32, (LSPAN * BATCH) / 128), 256, 0, stream>>>(
        long_x, (long long)GSTEP * BATCH * D, D,
        rs_w1, D, H, nullptr, hctx, nullptr, hbuf, 1);
    weights_kernel<<<BATCH, 32, 0, stream>>>(hbuf, rs_w2, rs_b2, dec_g, wg);

    // 8) weighted pooling + span scaling -> combined [128, 2048]
    pool_combine_kernel<<<(BATCH * D / 4) / 256, 256, 0, stream>>>(long_x, wl, wg, sw, combined);

    // 9) fused = gelu(combined @ fu_w + fu_b)       M=128 K=2048 N=1024
    wmma_gemm_bf16<<<dim3(D / 32, 1), 256, 0, stream>>>(
        combined, 0LL, 2 * D, fu_w, 2 * D, D, fu_b, nullptr, nullptr, fusedbuf, 1);

    // 10) adaptive = fused * sigmoid(fused @ gate_w + gate_b) -> out tail
    wmma_gemm_bf16<<<dim3(D / 32, 1), 256, 0, stream>>>(
        fusedbuf, 0LL, D, gate_w, D, D, gate_b, nullptr, fusedbuf,
        out + (long long)SEQ * BATCH, 2);
}
